// MaskFilter_1563368096170
// MI455X (gfx1250) — matrix operand order repaired and checked by reading, not compiled, not run on hardware
//
#include <hip/hip_runtime.h>
#include <hip/hip_bf16.h>
#include <math.h>

typedef __attribute__((ext_vector_type(16))) _Float16 v16h;
typedef __attribute__((ext_vector_type(8)))  float    v8f;

#define B_   16
#define C_   37
#define H_   224
#define W_   224
#define HW   (H_*W_)      /* 50176 */
#define WPR  7            /* 32-bit words per row */
#define NW   (H_*WPR)     /* 1568 words per image bitmask */

__device__ __forceinline__ int refl224(int p) {            // BORDER_REFLECT_101
    p = (p < 0) ? -p : p;
    return (p >= W_) ? (2*W_ - 2 - p) : p;
}
// Gaussian taps [1,4,6,4,1]/16 without a scratch-indexed array
__device__ __forceinline__ float wgt(int d) {
    if (d < 0 || d > 4) return 0.0f;
    return (d == 2) ? 0.375f : ((d == 1 || d == 3) ? 0.25f : 0.0625f);
}

// ---------------- K0: zero per-image max (ws is poisoned) ----------------
__global__ void k_init_max(int* maxv) {
    if (threadIdx.x < B_) maxv[threadIdx.x] = 0;
}

// ---------------- K1: channel argmax + per-image max of label ----------------
__global__ void k_argmax(const float* __restrict__ in, unsigned char* __restrict__ lab,
                         int* __restrict__ maxv) {
    int b = blockIdx.x / 196;                       // 196 blocks per image (50176/256)
    int p = (blockIdx.x % 196) * 256 + threadIdx.x;
    const float* base = in + (size_t)b * C_ * HW + p;
    float best = base[0];
    int bi = 0;
    for (int c = 1; c < C_; ++c) {
        float v = base[(size_t)c * HW];             // coalesced across lanes
        if (v > best) { best = v; bi = c; }         // first-max tie rule (strict >)
    }
    lab[(size_t)b * HW + p] = (unsigned char)bi;

    __shared__ int smax[256];
    smax[threadIdx.x] = bi;
    __syncthreads();
    for (int s = 128; s > 0; s >>= 1) {
        if (threadIdx.x < s) smax[threadIdx.x] = max(smax[threadIdx.x], smax[threadIdx.x + s]);
        __syncthreads();
    }
    if (threadIdx.x == 0) atomicMax(&maxv[b], smax[0]);
}

// ---------------- K2: q = floor(255 * lab / maxv), in place ----------------
__global__ void k_quant(unsigned char* __restrict__ q, const int* __restrict__ maxv) {
    int idx = blockIdx.x * 256 + threadIdx.x;
    int b = idx / HW;
    int mv = maxv[b];
    float norm = (mv > 0) ? ((float)q[idx] / (float)mv) : 0.0f;
    q[idx] = (unsigned char)floorf(255.0f * norm);
}

// ---------------- K3: separable 5x5 blur via WMMA, threshold > 0 ----------------
// One wave per 16x16 output tile. 3x v_wmma_f32_16x16x32_f16 per tile.
__global__ void k_blur_wmma(const unsigned char* __restrict__ q,
                            unsigned char* __restrict__ mask) {
    __shared__ _Float16 sh[8][32][16];              // per-wave H staging (f16)
    int lane = threadIdx.x & 31;
    int w    = threadIdx.x >> 5;
    int tile = blockIdx.x * 8 + w;                  // 392 blocks * 8 waves = 3136 tiles
    int b    = tile / 196;
    int t    = tile % 196;
    int row0 = (t / 14) * 16;
    int col0 = (t % 14) * 16;
    int n    = lane & 15;
    int half = lane >> 4;

    // B-matrix (32x16): horizontal band, B[k][n] = k5[k-n-6]
    v16h Bw;
    #pragma unroll
    for (int j = 0; j < 8; ++j) {
        int k0 = (j < 4) ? (2*j + 8*half) : (16 + 2*(j-4) + 8*half);   // B layout: same K order as A
        Bw[2*j]     = (_Float16)wgt(k0     - n - 6);
        Bw[2*j + 1] = (_Float16)wgt(k0 + 1 - n - 6);
    }

    const unsigned char* qb = q + (size_t)b * HW;

    // Stage 1: horizontal blur of rows [row0-8, row0+24)
    #pragma unroll
    for (int sub = 0; sub < 2; ++sub) {
        int rr = refl224(row0 - 8 + sub * 16 + (lane & 15));   // A: M = lane&15
        const unsigned char* qrow = qb + rr * W_;
        v16h A;
        #pragma unroll
        for (int i = 0; i < 8; ++i) {
            // A layout: VGPR0-3 -> K 0-7 / 8-15 by lane half; VGPR4-7 -> 16-23 / 24-31
            int kb = (i < 4) ? (2*i + 8*half) : (16 + 2*(i-4) + 8*half);
            A[2*i]     = (_Float16)(float)qrow[refl224(col0 - 8 + kb)];
            A[2*i + 1] = (_Float16)(float)qrow[refl224(col0 - 7 + kb)];
        }
        v8f c = {};
        c = __builtin_amdgcn_wmma_f32_16x16x32_f16(false, A, false, Bw,
                                                   (short)0, c, false, false);
        #pragma unroll
        for (int v = 0; v < 8; ++v) {
            int rloc = sub * 16 + v + 8 * half;     // C/D: M = v (+8 upper lanes)
            sh[w][rloc][n] = (_Float16)c[v];
        }
    }
    __syncthreads();

    // Stage 2: vertical blur. A = band (16x32), B = H (32x16) from LDS.
    v16h Av, Bh;
    int m = lane & 15;
    #pragma unroll
    for (int i = 0; i < 8; ++i) {
        int kb = (i < 4) ? (2*i + 8*half) : (16 + 2*(i-4) + 8*half);
        Av[2*i]     = (_Float16)wgt(kb     - m - 6);
        Av[2*i + 1] = (_Float16)wgt(kb + 1 - m - 6);
    }
    #pragma unroll
    for (int j = 0; j < 8; ++j) {
        int k0 = (j < 4) ? (2*j + 8*half) : (16 + 2*(j-4) + 8*half);
        Bh[2*j]     = sh[w][k0][n];
        Bh[2*j + 1] = sh[w][k0 + 1][n];
    }
    v8f o = {};
    o = __builtin_amdgcn_wmma_f32_16x16x32_f16(false, Av, false, Bh,
                                               (short)0, o, false, false);

    unsigned char* mb = mask + (size_t)b * HW;
    #pragma unroll
    for (int v = 0; v < 8; ++v) {
        int r  = row0 + v + 8 * half;
        int cc = col0 + n;
        mb[r * W_ + cc] = (rintf(o[v]) > 0.0f) ? 1 : 0;   // round-half-even, th=0
    }
}

// ---------------- K4/K5: binary opening (cross footprint) ----------------
__global__ void k_erode(const unsigned char* __restrict__ in, unsigned char* __restrict__ out) {
    int idx = blockIdx.x * 256 + threadIdx.x;
    int b = idx / HW, p = idx % HW;
    int r = p / W_, c = p % W_;
    const unsigned char* m = in + (size_t)b * HW;
    unsigned char v = m[p];                          // OOB treated as True
    v &= (r > 0)      ? m[p - W_] : 1;
    v &= (r < H_ - 1) ? m[p + W_] : 1;
    v &= (c > 0)      ? m[p - 1]  : 1;
    v &= (c < W_ - 1) ? m[p + 1]  : 1;
    out[idx] = v;
}
__global__ void k_dilate(const unsigned char* __restrict__ in, unsigned char* __restrict__ out) {
    int idx = blockIdx.x * 256 + threadIdx.x;
    int b = idx / HW, p = idx % HW;
    int r = p / W_, c = p % W_;
    const unsigned char* m = in + (size_t)b * HW;
    unsigned char v = m[p];                          // OOB treated as False
    v |= (r > 0)      ? m[p - W_] : 0;
    v |= (r < H_ - 1) ? m[p + W_] : 0;
    v |= (c > 0)      ? m[p - 1]  : 0;
    v |= (c < W_ - 1) ? m[p + 1]  : 0;
    out[idx] = v;
}

// ---------------- K6: hole fill (bitwise flood fill in LDS) + 3-ch output ----------------
__global__ void k_fill_out(const unsigned char* __restrict__ mask, float* __restrict__ out) {
    __shared__ unsigned comp[NW], scur[NW], snxt[NW];
    __shared__ int changed;
    int b = blockIdx.x, tid = threadIdx.x;
    const unsigned char* mb = mask + (size_t)b * HW;

    for (int wd = tid; wd < NW; wd += 256) {
        int r = wd / WPR, c = wd % WPR;
        const unsigned char* row = mb + r * W_ + c * 32;
        unsigned bits = 0;
        #pragma unroll
        for (int j = 0; j < 32; ++j) bits |= (row[j] ? 0u : 1u) << j;   // comp = ~m
        comp[wd] = bits;
        unsigned bm = (r == 0 || r == H_ - 1) ? 0xFFFFFFFFu
                    : ((c == 0 ? 1u : 0u) | (c == WPR - 1 ? 0x80000000u : 0u));
        scur[wd] = bits & bm;                        // seed = comp & border
    }
    __syncthreads();

    for (int iter = 0; iter < 4096; ++iter) {
        if (tid == 0) changed = 0;
        __syncthreads();
        int ch = 0;
        for (int wd = tid; wd < NW; wd += 256) {
            int r = wd / WPR, c = wd % WPR;
            unsigned val = scur[wd];
            unsigned nb  = val;
            nb |= (val << 1) | ((c > 0)       ? (scur[wd - 1] >> 31) : 0u);
            nb |= (val >> 1) | ((c < WPR - 1) ? (scur[wd + 1] << 31) : 0u);
            nb |= (r > 0)      ? scur[wd - WPR] : 0u;
            nb |= (r < H_ - 1) ? scur[wd + WPR] : 0u;
            unsigned ns = comp[wd] & nb;             // dilate constrained to comp
            snxt[wd] = ns;
            ch |= (ns != val) ? 1 : 0;
        }
        if (ch) atomicOr(&changed, 1);
        __syncthreads();
        int cont = changed;
        for (int wd = tid; wd < NW; wd += 256) scur[wd] = snxt[wd];
        __syncthreads();
        if (!cont) break;                            // uniform exit
    }

    float* ob = out + (size_t)b * 3 * HW;
    for (int p = tid; p < HW; p += 256) {
        int r = p / W_, c = p % W_;
        unsigned bg = (scur[r * WPR + (c >> 5)] >> (c & 31)) & 1u;
        float v = bg ? 0.0f : 1.0f;                  // foreground = ~background
        ob[p] = v; ob[HW + p] = v; ob[2 * HW + p] = v;
    }
}

extern "C" void kernel_launch(void* const* d_in, const int* in_sizes, int n_in,
                              void* d_out, int out_size, void* d_ws, size_t ws_size,
                              hipStream_t stream) {
    (void)in_sizes; (void)n_in; (void)out_size; (void)ws_size;
    const float* x = (const float*)d_in[0];          // (16,37,224,224) fp32; label==1 path
    unsigned char* q  = (unsigned char*)d_ws;                    // lab -> q  (802816 B)
    unsigned char* mA = q  + (size_t)B_ * HW;                    // mask A    (802816 B)
    unsigned char* mB = mA + (size_t)B_ * HW;                    // mask B    (802816 B)
    int* maxv         = (int*)(mB + (size_t)B_ * HW);            // 16 ints
    float* out = (float*)d_out;

    hipLaunchKernelGGL(k_init_max,  dim3(1),          dim3(32),  0, stream, maxv);
    hipLaunchKernelGGL(k_argmax,    dim3(196 * B_),   dim3(256), 0, stream, x, q, maxv);
    hipLaunchKernelGGL(k_quant,     dim3(B_*HW/256),  dim3(256), 0, stream, q, maxv);
    hipLaunchKernelGGL(k_blur_wmma, dim3(B_*196/8),   dim3(256), 0, stream, q, mA);
    hipLaunchKernelGGL(k_erode,     dim3(B_*HW/256),  dim3(256), 0, stream, mA, mB);
    hipLaunchKernelGGL(k_dilate,    dim3(B_*HW/256),  dim3(256), 0, stream, mB, mA);
    hipLaunchKernelGGL(k_fill_out,  dim3(B_),         dim3(256), 0, stream, mA, out);
}
